// AsyNonLocal2D_27857157882106
// MI455X (gfx1250) — matrix operand order repaired and checked by reading, not compiled, not run on hardware
//
#include <hip/hip_runtime.h>
#include <hip/hip_bf16.h>

// ---------------------------------------------------------------------------
// AsyNonLocal2D for MI455X (gfx1250, wave32, WMMA + TDM).
// B=4, Cin=256, I=128, H=W=64, N=4096.
//   1) proj_kernel   : theta/phi ([b][n][I] bf16), g ([b][I][n] bf16)
//   2) attn_kernel   : fused flash-attention; key slabs staged block-wide in
//                      LDS by the Tensor Data Mover (tensor_load_to_lds,
//                      double-buffered, s_wait_tensorcnt), WMMA bf16 matmuls,
//                      permlane16 softmax reductions.
//   3) outproj_kernel: out = querry + Wout x y + bout (WMMA)
// Workspace: theta 4MB | phi 4MB | g 4MB | y 4MB = 16MB.
// ---------------------------------------------------------------------------

#define BB   4
#define CIN  256
#define II   128
#define NN   4096

typedef __bf16 bf16;
typedef bf16  v16bf __attribute__((ext_vector_type(16)));
typedef bf16  v8bf  __attribute__((ext_vector_type(8)));
typedef float v8f   __attribute__((ext_vector_type(8)));
typedef unsigned int v4u __attribute__((ext_vector_type(4)));
typedef int   v4i  __attribute__((ext_vector_type(4)));
typedef int   v8i  __attribute__((ext_vector_type(8)));

__device__ __forceinline__ unsigned short f2bfbits(float f) {
    unsigned u = __builtin_bit_cast(unsigned, f);
    u += 0x7FFFu + ((u >> 16) & 1u);          // round-to-nearest-even
    return (unsigned short)(u >> 16);
}
__device__ __forceinline__ bf16 f2bf(float f) {
    return __builtin_bit_cast(bf16, f2bfbits(f));
}

// A-fragment (16x32, M x K): lane holds row (lane&15); lanes<16 carry K 0..7 &
// 16..23, lanes>=16 carry K 8..15 & 24..31 — two contiguous 16B chunks.
__device__ __forceinline__ v16bf load_mat_a(const bf16* p, int ld) {
    int lane = threadIdx.x & 31;
    const bf16* q = p + (size_t)(lane & 15) * ld + ((lane & 16) ? 8 : 0);
    v8bf lo = *(const v8bf*)q;
    v8bf hi = *(const v8bf*)(q + 16);
    return __builtin_shufflevector(lo, hi, 0,1,2,3,4,5,6,7,8,9,10,11,12,13,14,15);
}

// B-fragment (32x16, K x N): lane holds column (lane&15); lanes<16 carry
// K 0..7 & 16..23, lanes>=16 carry K 8..15 & 24..31 — two contiguous 16B chunks.
__device__ __forceinline__ v16bf load_mat_b(const bf16* p, int ld) {
    int lane = threadIdx.x & 31;
    const bf16* q = p + (size_t)(lane & 15) * ld + ((lane & 16) ? 8 : 0);
    v8bf lo = *(const v8bf*)q;
    v8bf hi = *(const v8bf*)(q + 16);
    return __builtin_shufflevector(lo, hi, 0,1,2,3,4,5,6,7,8,9,10,11,12,13,14,15);
}

__device__ __forceinline__ v8f wmma_bf16(v16bf a, v16bf b, v8f c) {
    return __builtin_amdgcn_wmma_f32_16x16x32_bf16(
        false, a, false, b, (short)0, c, false, false);
}

// ---- permlane16-based XOR reductions within each 16-lane row (pure VALU) ---
__device__ __forceinline__ float permx(float v, int lo, int hi) {
    int i = __builtin_bit_cast(int, v);
    int r = __builtin_amdgcn_permlane16(i, i, lo, hi, false, false);
    return __builtin_bit_cast(float, r);
}
__device__ __forceinline__ float redmax16(float v) {
    v = fmaxf(v, permx(v, 0x67452301, 0xEFCDAB89));   // xor 1
    v = fmaxf(v, permx(v, 0x54761032, 0xDCFE98BA));   // xor 2
    v = fmaxf(v, permx(v, 0x32107654, 0xBA98FEDC));   // xor 4
    v = fmaxf(v, permx(v, 0xFEDCBA98, 0x76543210));   // xor 8
    return v;
}
__device__ __forceinline__ float redsum16(float v) {
    v += permx(v, 0x67452301, 0xEFCDAB89);
    v += permx(v, 0x54761032, 0xDCFE98BA);
    v += permx(v, 0x32107654, 0xBA98FEDC);
    v += permx(v, 0xFEDCBA98, 0x76543210);
    return v;
}

// ---- Tensor Data Mover: async 2D tile (bf16) global -> LDS -----------------
__device__ __forceinline__ unsigned ldsoff(const void* p) {
    return (unsigned)(unsigned long long)p;   // flat shared addr low 32b = LDS byte offset
}
__device__ __forceinline__ void tdm_load_2d(unsigned lds_byte_addr, const void* gaddr,
                                            unsigned tensor_d0, unsigned tensor_d1,
                                            unsigned tile_d0, unsigned tile_d1,
                                            unsigned long long stride0_elems)
{
    unsigned long long ga = (unsigned long long)gaddr;
    v4u g0;
    g0[0] = 1u;                                              // count=1, user desc
    g0[1] = lds_byte_addr;                                   // lds_addr
    g0[2] = (unsigned)(ga & 0xFFFFFFFFu);                    // global_addr lo
    g0[3] = (unsigned)((ga >> 32) & 0x01FFFFFFu) | (2u << 30); // addr hi | type=2
    v8i g1;
    g1[0] = (int)(1u << 16);                                 // data_size=1 (2B)
    g1[1] = (int)((tensor_d0 & 0xFFFFu) << 16);              // tensor_dim0 lo16
    g1[2] = (int)((tensor_d0 >> 16) | ((tensor_d1 & 0xFFFFu) << 16));
    g1[3] = (int)((tensor_d1 >> 16) | (tile_d0 << 16));      // | tile_dim0
    g1[4] = (int)tile_d1;                                    // tile_dim1, tile_dim2=0
    g1[5] = (int)(stride0_elems & 0xFFFFFFFFu);              // dim0 stride lo
    g1[6] = (int)((stride0_elems >> 32) & 0xFFFFu);          // stride hi | dim1 stride=0
    g1[7] = 0;
    v4i z4 = {0, 0, 0, 0};
#if defined(__clang_major__) && __clang_major__ >= 23
    v8i z8 = {0, 0, 0, 0, 0, 0, 0, 0};
    __builtin_amdgcn_tensor_load_to_lds(g0, g1, z4, z4, z8, 0);
#else
    __builtin_amdgcn_tensor_load_to_lds(g0, g1, z4, z4, 0);
#endif
}

// ---------------------------------------------------------------------------
// Stage 1: 1x1-conv projections with bf16 output, layouts tuned for WMMA.
// ---------------------------------------------------------------------------
__global__ __launch_bounds__(256) void proj_kernel(
    const float* __restrict__ querry, const float* __restrict__ refr,
    const float* __restrict__ Wg, const float* __restrict__ bg,
    const float* __restrict__ Wt, const float* __restrict__ bt,
    const float* __restrict__ Wp, const float* __restrict__ bp,
    bf16* __restrict__ theta, bf16* __restrict__ phi, bf16* __restrict__ g)
{
    const int p  = blockIdx.z;
    const int b  = blockIdx.y;
    const int n0 = blockIdx.x * 16;

    const float* x    = (p == 0) ? querry : refr;
    const float* W    = (p == 0) ? Wt : ((p == 1) ? Wp : Wg);
    const float* bias = (p == 0) ? bt : ((p == 1) ? bp : bg);

    __shared__ float xs[CIN * 16];                 // 16 KB: x[:, n0..n0+15]
    for (int idx = threadIdx.x; idx < CIN * 16; idx += 256) {
        int c = idx >> 4, nl = idx & 15;
        xs[idx] = x[((size_t)b * CIN + c) * NN + n0 + nl];
    }
    __syncthreads();

    const int i   = threadIdx.x & 127;
    const int nl0 = (threadIdx.x >> 7) * 8;

    float acc[8];
    #pragma unroll
    for (int r = 0; r < 8; ++r) acc[r] = bias[i];

    const float* wrow = W + (size_t)i * CIN;
    for (int c = 0; c < CIN; ++c) {
        float w = wrow[c];
        #pragma unroll
        for (int r = 0; r < 8; ++r) acc[r] = fmaf(w, xs[c * 16 + nl0 + r], acc[r]);
    }

    #pragma unroll
    for (int r = 0; r < 8; ++r) {
        int n = n0 + nl0 + r;
        bf16 v = f2bf(acc[r]);
        if (p == 2) g[((size_t)b * II + i) * NN + n] = v;
        else {
            bf16* dst = (p == 0) ? theta : phi;
            dst[((size_t)b * NN + n) * II + i] = v;
        }
    }
}

// ---------------------------------------------------------------------------
// Stage 2: fused flash attention.
//  - Block = 8 waves, each owning a 16-query tile (q0..q0+15) x full I=128.
//  - Key slabs (32 keys): phi[32x128] and g[128x32] bf16, TDM-staged into
//    double-buffered LDS by wave 0; all waves consume via ds_load fragments.
//  - Per 32-key step: 8 WMMA for S, permlane16 online softmax, 8 WMMA for O.
// ---------------------------------------------------------------------------
__global__ __launch_bounds__(256) void attn_kernel(
    const bf16* __restrict__ thetaT, const bf16* __restrict__ phiT,
    const bf16* __restrict__ gI, bf16* __restrict__ y)
{
    __shared__ bf16 phs[2][32 * II];               // 2 x 8 KB  phi slabs
    __shared__ bf16 gs [2][II * 32];               // 2 x 8 KB  g   slabs
    __shared__ unsigned short pbuf[8 * 16 * 32];   // per-wave P restage (8 KB)

    const int lane = threadIdx.x & 31;
    const int wave = threadIdx.x >> 5;
    const int b    = blockIdx.y;
    const int q0   = blockIdx.x * 128 + wave * 16;

    const bf16* th = thetaT + ((size_t)b * NN + q0) * II;
    const bf16* ph = phiT   + (size_t)b * NN * II;
    const bf16* gv = gI     + (size_t)b * II * NN;
    unsigned short* P = pbuf + wave * (16 * 32);

    // Persistent A fragments: theta rows q0..q0+15, K=I in 4 steps of 32.
    v16bf A[4];
    #pragma unroll
    for (int kk = 0; kk < 4; ++kk) A[kk] = load_mat_a(th + kk * 32, II);

    v8f O[8];
    float m[8], l[8];
    #pragma unroll
    for (int j = 0; j < 8; ++j) {
        #pragma unroll
        for (int t = 0; t < 8; ++t) O[j][t] = 0.0f;
        m[j] = -__builtin_inff();
        l[j] = 0.0f;
    }

    // Prologue: stage slab 0, wait, publish.
    if (wave == 0) {
        tdm_load_2d(ldsoff(phs[0]), ph, II, NN, II, 32, II);
        tdm_load_2d(ldsoff(gs[0]),  gv, NN, II, 32, II, NN);
    }
    __builtin_amdgcn_s_wait_tensorcnt(0);
    __syncthreads();

    const float scale = 0.08838834764831845f;      // 1/sqrt(128)
    const int   half  = (lane & 16) ? 8 : 0;

    for (int r0 = 0; r0 < NN; r0 += 32) {
        const int cur = (r0 >> 5) & 1;
        // Kick TDM for the next slab; overlaps with this step's compute.
        if (wave == 0 && (r0 + 32) < NN) {
            const int nxt = cur ^ 1;
            tdm_load_2d(ldsoff(phs[nxt]), ph + (size_t)(r0 + 32) * II, II, NN, II, 32, II);
            tdm_load_2d(ldsoff(gs[nxt]),  gv + (r0 + 32),              NN, II, 32, II, NN);
        }

        // ---- S = theta x phi^T over the 32-key slab (from LDS) ----
        v8f S0, S1;
        #pragma unroll
        for (int t = 0; t < 8; ++t) { S0[t] = 0.0f; S1[t] = 0.0f; }
        #pragma unroll
        for (int kk = 0; kk < 4; ++kk) {
            v16bf B0 = load_mat_b(phs[cur] + kk * 32, II);            // keys 0..15
            v16bf B1 = load_mat_b(phs[cur] + 16 * II + kk * 32, II);  // keys 16..31
            S0 = wmma_bf16(A[kk], B0, S0);
            S1 = wmma_bf16(A[kk], B1, S1);
        }

        // ---- online softmax (row reductions via v_permlane16) ----
        float alpha[8];
        #pragma unroll
        for (int j = 0; j < 8; ++j) {
            float s0 = S0[j] * scale;
            float s1 = S1[j] * scale;
            float mx = redmax16(fmaxf(s0, s1));
            float mn = fmaxf(m[j], mx);
            alpha[j] = __expf(m[j] - mn);
            float p0 = __expf(s0 - mn);
            float p1 = __expf(s1 - mn);
            l[j] = l[j] * alpha[j] + redsum16(p0 + p1);
            m[j] = mn;
            int row = j + half;
            P[row * 32 +      (lane & 15)] = f2bfbits(p0);
            P[row * 32 + 16 + (lane & 15)] = f2bfbits(p1);
        }
        #pragma unroll
        for (int iblk = 0; iblk < 8; ++iblk)
            #pragma unroll
            for (int j = 0; j < 8; ++j) O[iblk][j] *= alpha[j];

        // same-wave DS ordering: drain P stores before re-reading as A-frag
        asm volatile("s_wait_dscnt 0" ::: "memory");
        v16bf Pa = load_mat_a((const bf16*)P, 32);

        // ---- O += P x g (8 blocks of 16 output channels, g from LDS) ----
        #pragma unroll
        for (int iblk = 0; iblk < 8; ++iblk) {
            v16bf Bg = load_mat_b(gs[cur] + (iblk * 16) * 32, 32);
            O[iblk] = wmma_bf16(Pa, Bg, O[iblk]);
        }

        // Next slab must be resident before anyone reads it; also fences the
        // slab we just finished reading before wave0 overwrites it.
        __builtin_amdgcn_s_wait_tensorcnt(0);
        __syncthreads();
    }

    // ---- epilogue: y[b][q][i] = O / l (bf16) ----
    #pragma unroll
    for (int iblk = 0; iblk < 8; ++iblk) {
        #pragma unroll
        for (int j = 0; j < 8; ++j) {
            int row = q0 + j + half;
            int col = iblk * 16 + (lane & 15);
            y[((size_t)b * NN + row) * II + col] = f2bf(O[iblk][j] / l[j]);
        }
    }
}

// ---------------------------------------------------------------------------
// Stage 3: out = querry + Wout x y + bout   (WMMA, residual fused at store)
// ---------------------------------------------------------------------------
__global__ __launch_bounds__(256) void outproj_kernel(
    const bf16* __restrict__ y, const float* __restrict__ Wout,
    const float* __restrict__ bout, const float* __restrict__ querry,
    float* __restrict__ out)
{
    const int lane = threadIdx.x & 31;
    const int wave = threadIdx.x >> 5;
    const int b    = blockIdx.z;
    const int o0   = (blockIdx.y * 8 + wave) * 16;
    const int n0   = blockIdx.x * 16;

    v16bf Aw[4];
    {
        const int mm = lane & 15, koff = (lane & 16) ? 8 : 0;
        #pragma unroll
        for (int kk = 0; kk < 4; ++kk) {
            const float* q = Wout + (size_t)(o0 + mm) * II + kk * 32 + koff;
            v16bf a;
            #pragma unroll
            for (int t = 0; t < 8; ++t) {
                a[t]     = f2bf(q[t]);
                a[t + 8] = f2bf(q[t + 16]);
            }
            Aw[kk] = a;
        }
    }

    v8f C;
    #pragma unroll
    for (int t = 0; t < 8; ++t) C[t] = 0.0f;

    const bf16* yb = y + ((size_t)b * NN + n0) * II;
    #pragma unroll
    for (int kk = 0; kk < 4; ++kk)
        C = wmma_bf16(Aw[kk], load_mat_b(yb + kk * 32, II), C);

    const int half = (lane & 16) ? 8 : 0;
    #pragma unroll
    for (int j = 0; j < 8; ++j) {
        int o = o0 + j + half;
        int n = n0 + (lane & 15);
        size_t idx = ((size_t)b * CIN + o) * NN + n;
        out[idx] = querry[idx] + C[j] + bout[o];
    }
}

// ---------------------------------------------------------------------------
extern "C" void kernel_launch(void* const* d_in, const int* in_sizes, int n_in,
                              void* d_out, int out_size, void* d_ws, size_t ws_size,
                              hipStream_t stream)
{
    const float* querry = (const float*)d_in[0];
    const float* refr   = (const float*)d_in[1];
    const float* Wg     = (const float*)d_in[2];
    const float* bg     = (const float*)d_in[3];
    const float* Wt     = (const float*)d_in[4];
    const float* bt     = (const float*)d_in[5];
    const float* Wp     = (const float*)d_in[6];
    const float* bp     = (const float*)d_in[7];
    const float* Wout   = (const float*)d_in[8];
    const float* bout   = (const float*)d_in[9];
    float* out = (float*)d_out;

    const size_t SEG = (size_t)BB * NN * II * sizeof(bf16);   // 4 MB
    char* wsb = (char*)d_ws;
    bf16* theta = (bf16*)(wsb + 0 * SEG);
    bf16* phi   = (bf16*)(wsb + 1 * SEG);
    bf16* g     = (bf16*)(wsb + 2 * SEG);
    bf16* yy    = (bf16*)(wsb + 3 * SEG);

    proj_kernel<<<dim3(NN / 16, BB, 3), 256, 0, stream>>>(
        querry, refr, Wg, bg, Wt, bt, Wp, bp, theta, phi, g);

    attn_kernel<<<dim3(NN / 128, BB), 256, 0, stream>>>(theta, phi, g, yy);

    outproj_kernel<<<dim3(NN / 16, CIN / 128, BB), 256, 0, stream>>>(
        yy, Wout, bout, querry, out);
}
